// GATNet_82197084111150
// MI455X (gfx1250) — matrix operand order repaired and checked by reading, not compiled, not run on hardware
//
#include <hip/hip_runtime.h>
#include <hip/hip_bf16.h>
#include <stdint.h>
#include <stddef.h>

// ---------------------------------------------------------------------------
// GAT 2-layer pipeline for MI455X (gfx1250).
// Dense GEMMs via v_wmma_f32_16x16x32_bf16 with double-buffered
// global_load_async_to_lds_b128 staging (ASYNCcnt pipeline); edge phase via
// coalesced wave-per-edge gather + global float atomics + prefetch.
// ---------------------------------------------------------------------------

#define NNODES 100000
#define NEDGES 800000
#define NEG_SLOPE 0.2f

typedef __attribute__((ext_vector_type(16))) __bf16 v16bf;
typedef __attribute__((ext_vector_type(8)))  float  v8f;

union FragBF { v16bf v; unsigned int u[8]; };

__device__ __forceinline__ unsigned short f32_to_bf16(float f) {
  unsigned int u = __float_as_uint(f);
  u += 0x7FFFu + ((u >> 16) & 1u);       // round-to-nearest-even
  return (unsigned short)(u >> 16);
}

// ------------------- CDNA5 async-load-to-LDS primitives --------------------

// LDS offset of a __shared__ object: flat LDS addresses carry the byte offset
// in the low 32 bits (aperture in the high bits), so truncation is the offset.
__device__ __forceinline__ unsigned lds_addr32(const void* p) {
  return (unsigned)(uintptr_t)p;
}

// 16B global -> LDS, no VGPR round trip; tracked by ASYNCcnt.
__device__ __forceinline__ void async_load_b128(unsigned lds_off, const void* gaddr) {
  asm volatile("global_load_async_to_lds_b128 %0, %1, off"
               :: "v"(lds_off), "v"(gaddr)
               : "memory");
}

__device__ __forceinline__ void wait_async_le3() {
  asm volatile("s_wait_asynccnt 0x3" ::: "memory");
}
__device__ __forceinline__ void wait_async_le0() {
  asm volatile("s_wait_asynccnt 0x0" ::: "memory");
}

// --------------------------- conversion / packing ---------------------------

__global__ void cvt_f32_bf16_kernel(const float* __restrict__ in,
                                    unsigned short* __restrict__ out, int n) {
  int i = blockIdx.x * blockDim.x + threadIdx.x;
  if (i < n) out[i] = f32_to_bf16(in[i]);
}

// Pack W [K, Nout] f32 row-major -> Wp [Nout][K/2] uint (bf16 pair along K).
__global__ void pack_w_kernel(const float* __restrict__ W,
                              unsigned int* __restrict__ Wp, int K, int Nout) {
  int i = blockIdx.x * blockDim.x + threadIdx.x;
  int Kp = K >> 1;
  if (i >= Nout * Kp) return;
  int n = i / Kp, p = i - n * Kp;
  unsigned int lo = f32_to_bf16(W[(size_t)(2 * p)     * Nout + n]);
  unsigned int hi = f32_to_bf16(W[(size_t)(2 * p + 1) * Nout + n]);
  Wp[i] = lo | (hi << 16);
}

// ------------------------------- WMMA GEMM ---------------------------------
// D[M,Nout] = A[M,K] (bf16, row-major, pair-packed uints) x B (pre-packed
// [Nout][K/2]).  Block = 256 threads = 8 waves; tile 128(M) x 64(N); each
// wave owns a 32x32 patch = 2x2 WMMA 16x16x32 tiles.  LDS is double-buffered
// and filled with async loads: each thread issues 3x b128 per K-tile.

#define LDS_A_UINTS (128 * 16)   // 8 KB per buffer
#define LDS_B_UINTS (64 * 16)    // 4 KB per buffer

__device__ __forceinline__ void stage_tile_async(
    const unsigned int* __restrict__ A, const unsigned int* __restrict__ Bp,
    unsigned ldsA, unsigned ldsB,
    int t, int m0, int n0, int M, int Kp, int k0p) {
  // A tile: 128 rows x 16 pairs = 512 uint4; 2 per thread.
#pragma unroll
  for (int i = 0; i < 2; ++i) {
    int idx4 = t + i * 256;
    int r = idx4 >> 2, p4 = idx4 & 3;
    int gr = m0 + r; if (gr >= M) gr = M - 1;          // clamp (store guarded)
    async_load_b128(ldsA + idx4 * 16,
                    A + (size_t)gr * Kp + k0p + p4 * 4);
  }
  // B tile: 64 cols x 16 pairs = 256 uint4; 1 per thread.
  {
    int n = t >> 2, p4 = t & 3;
    async_load_b128(ldsB + t * 16,
                    Bp + (size_t)(n0 + n) * Kp + k0p + p4 * 4);
  }
}

__global__ void __launch_bounds__(256)
gemm_bf16_wmma_kernel(const unsigned int* __restrict__ A,   // [M][K/2]
                      const unsigned int* __restrict__ Bp,  // [Nout][K/2]
                      float* __restrict__ D,
                      int M, int K, int Nout) {
  __shared__ unsigned int lA[2][LDS_A_UINTS];
  __shared__ unsigned int lB[2][LDS_B_UINTS];

  const int t    = threadIdx.x;
  const int wave = t >> 5;
  const int lane = t & 31;
  const int wy   = wave >> 1;            // 0..3 -> 32-row band
  const int wx   = wave & 1;             // 0..1 -> 32-col band
  const int l16  = lane & 15;
  const int kh   = lane >> 4;            // lane half (K-phase select)
  const int m0   = blockIdx.x * 128;
  const int n0   = blockIdx.y * 64;
  const int Kp   = K >> 1;
  const int T    = Kp >> 4;              // K-tiles of 32

  const unsigned baseA = lds_addr32(&lA[0][0]);
  const unsigned baseB = lds_addr32(&lB[0][0]);

  v8f acc[2][2];
  acc[0][0] = v8f{}; acc[0][1] = v8f{}; acc[1][0] = v8f{}; acc[1][1] = v8f{};

  stage_tile_async(A, Bp, baseA, baseB, t, m0, n0, M, Kp, 0);

  int cur = 0;
  for (int it = 0; it < T; ++it) {
    if (it + 1 < T) {
      // fill the other buffer while consuming this one
      stage_tile_async(A, Bp,
                       baseA + (unsigned)((cur ^ 1) * LDS_A_UINTS * 4),
                       baseB + (unsigned)((cur ^ 1) * LDS_B_UINTS * 4),
                       t, m0, n0, M, Kp, (it + 1) * 16);
      wait_async_le3();   // in-order: <=3 outstanding => tile `it` resident
    } else {
      wait_async_le0();
    }
    __syncthreads();       // all threads' tile-`it` data visible

    const unsigned int* cA = &lA[cur][0];
    const unsigned int* cB = &lB[cur][0];

    // Build fragments per the ISA 16-bit A 16x32 / B 32x16 wave32 layouts.
    FragBF af[2], bf_[2];
#pragma unroll
    for (int ty = 0; ty < 2; ++ty) {
      int ml = wy * 32 + ty * 16 + l16;                  // row within tile
#pragma unroll
      for (int v = 0; v < 8; ++v) {
        int p = (v < 4 ? v : 8 + (v - 4)) + kh * 4;      // A k-pair index
        af[ty].u[v] = cA[ml * 16 + p];
      }
    }
#pragma unroll
    for (int tx = 0; tx < 2; ++tx) {
      int nl = wx * 32 + tx * 16 + l16;                  // col within tile
#pragma unroll
      for (int v = 0; v < 8; ++v)
        bf_[tx].u[v] = cB[nl * 16 + (v < 4 ? v : 8 + (v - 4)) + kh * 4];         // B k-pair index
    }

#pragma unroll
    for (int ty = 0; ty < 2; ++ty)
#pragma unroll
      for (int tx = 0; tx < 2; ++tx)
        acc[ty][tx] = __builtin_amdgcn_wmma_f32_16x16x32_bf16(
            false, af[ty].v, false, bf_[tx].v,
            (short)0, acc[ty][tx], false, false);

    __syncthreads();       // done reading buf `cur` before it is refilled
    cur ^= 1;
  }

  // Store: C/D layout — VGPR r: M = r + 8*(lane>=16), N = lane&15.
#pragma unroll
  for (int ty = 0; ty < 2; ++ty) {
#pragma unroll
    for (int tx = 0; tx < 2; ++tx) {
      int n = n0 + wx * 32 + tx * 16 + l16;
#pragma unroll
      for (int r = 0; r < 8; ++r) {
        int m = m0 + wy * 32 + ty * 16 + r + kh * 8;
        if (m < M) D[(size_t)m * Nout + n] = acc[ty][tx][r];
      }
    }
  }
}

// ------------------------- attention logits (dots) --------------------------
// One wave per node: alpha_src[n] = h[n,:].a_src, alpha_dst[n] = h[n,:].a_dst.

__global__ void rowdot2_kernel(const float* __restrict__ h,
                               const float* __restrict__ va,
                               const float* __restrict__ vb,
                               float* __restrict__ oa, float* __restrict__ ob,
                               int n_nodes, int C) {
  int wg   = (blockIdx.x * blockDim.x + threadIdx.x) >> 5;
  int lane = threadIdx.x & 31;
  if (wg >= n_nodes) return;
  const float* row = h + (size_t)wg * C;
  float sa = 0.f, sb = 0.f;
  for (int c = lane; c < C; c += 32) {
    float hv = row[c];
    sa += hv * va[c];
    sb += hv * vb[c];
  }
#pragma unroll
  for (int off = 16; off; off >>= 1) {
    sa += __shfl_xor(sa, off, 32);
    sb += __shfl_xor(sb, off, 32);
  }
  if (lane == 0) { oa[wg] = sa; ob[wg] = sb; }
}

// ------------------------------ init kernels -------------------------------

__global__ void fill_f32_kernel(float* __restrict__ p, float v, int n) {
  int i = blockIdx.x * blockDim.x + threadIdx.x;
  if (i < n) p[i] = v;
}

__global__ void bias_init_kernel(float* __restrict__ p,
                                 const float* __restrict__ bias,
                                 int total, int C) {
  int i = blockIdx.x * blockDim.x + threadIdx.x;
  if (i < total) p[i] = bias[i % C];
}

// ------------------------------ edge kernels -------------------------------

__device__ __forceinline__ void atomicMaxF(float* addr, float v) {
  // IEEE trick: positives via signed-int max, negatives via unsigned min.
  if (v >= 0.f) atomicMax((int*)addr, __float_as_int(v));
  else          atomicMin((unsigned int*)addr, __float_as_uint(v));
}

__device__ __forceinline__ void edge_sd(const int* __restrict__ src,
                                        const int* __restrict__ dst,
                                        int e, int E, int& s, int& d) {
  if (e < E) { s = src[e]; d = dst[e]; }
  else       { s = e - E; d = e - E; }       // self-loop
}

__global__ void edge_max_kernel(const int* __restrict__ src,
                                const int* __restrict__ dst,
                                const float* __restrict__ as_,
                                const float* __restrict__ ad_,
                                float* __restrict__ e_out,
                                float* __restrict__ emax, int E, int total) {
  int e = blockIdx.x * blockDim.x + threadIdx.x;
  if (e >= total) return;
  int s, d; edge_sd(src, dst, e, E, s, d);
  float v = as_[s] + ad_[d];
  v = (v > 0.f) ? v : NEG_SLOPE * v;         // LeakyReLU(0.2)
  e_out[e] = v;
  atomicMaxF(&emax[d], v);
}

__global__ void edge_expsum_kernel(const int* __restrict__ src,
                                   const int* __restrict__ dst,
                                   float* __restrict__ e_out,
                                   const float* __restrict__ emax,
                                   float* __restrict__ denom, int E, int total) {
  int e = blockIdx.x * blockDim.x + threadIdx.x;
  if (e >= total) return;
  int s, d; edge_sd(src, dst, e, E, s, d);
  float ee = __expf(e_out[e] - emax[d]);
  e_out[e] = ee;
  atomicAdd(&denom[d], ee);
}

// 4 edges per wave: out[dst,:] += h[src,:] * (ee / denom[dst]); lanes stride
// channels -> coalesced gather of h[src] and coalesced atomic adds.  Next
// edge's source row is prefetched while current row's atomics issue.
#define EDGES_PER_WAVE 4

__global__ void edge_agg_kernel(const int* __restrict__ src,
                                const int* __restrict__ dst,
                                const float* __restrict__ e_out,
                                const float* __restrict__ denom,
                                const float* __restrict__ h,
                                float* __restrict__ out,
                                int E, int total, int C) {
  int wv   = (blockIdx.x * blockDim.x + threadIdx.x) >> 5;
  int lane = threadIdx.x & 31;
  int e0   = wv * EDGES_PER_WAVE;
  if (e0 >= total) return;
#pragma unroll
  for (int k = 0; k < EDGES_PER_WAVE; ++k) {
    int e = e0 + k;
    if (e >= total) return;
    // prefetch next edge's source row (random 1KB/256B gather)
    if (k + 1 < EDGES_PER_WAVE && e + 1 < total) {
      int sn, dn; edge_sd(src, dst, e + 1, E, sn, dn);
      int cpf = lane * 8;
      if (cpf < C) __builtin_prefetch(h + (size_t)sn * C + cpf, 0, 0);
    }
    int s, d; edge_sd(src, dst, e, E, s, d);
    float w = e_out[e] / denom[d];
    const float* hs = h + (size_t)s * C;
    float*       od = out + (size_t)d * C;
    for (int c = lane; c < C; c += 32)
      atomicAdd(&od[c], hs[c] * w);
  }
}

// ReLU + convert to bf16 (layer-2 GEMM input). Input already includes bias.
__global__ void relu_bf16_kernel(const float* __restrict__ in,
                                 unsigned short* __restrict__ out, int n) {
  int i = blockIdx.x * blockDim.x + threadIdx.x;
  if (i < n) {
    float v = in[i];
    out[i] = f32_to_bf16(v > 0.f ? v : 0.f);
  }
}

// --------------------------------- driver ----------------------------------

static inline int cdiv(int a, int b) { return (a + b - 1) / b; }

extern "C" void kernel_launch(void* const* d_in, const int* in_sizes, int n_in,
                              void* d_out, int out_size, void* d_ws, size_t ws_size,
                              hipStream_t stream) {
  const float* x    = (const float*)d_in[0];   // [N,128]
  const int*   ei   = (const int*)  d_in[1];   // [2,E] flat
  const float* W1   = (const float*)d_in[2];   // [128,256]
  const float* a1s  = (const float*)d_in[3];
  const float* a1d  = (const float*)d_in[4];
  const float* b1   = (const float*)d_in[5];
  const float* W2   = (const float*)d_in[6];   // [256,64]
  const float* a2s  = (const float*)d_in[7];
  const float* a2d  = (const float*)d_in[8];
  const float* b2   = (const float*)d_in[9];
  float* out = (float*)d_out;                  // [N,64]

  const int N = NNODES, E = NEDGES, ET = NEDGES + NNODES;
  const int C1in = 128, C1 = 256, C2 = 64;
  const int* src = ei;
  const int* dst = ei + E;

  // Workspace carve-out (aligned 256B). h2 aliases xb (both 25.6 MB, xb dead
  // by the time h2 is produced).
  char* w = (char*)d_ws;
  size_t off = 0;
  auto take = [&](size_t bytes) { void* p = w + off; off += (bytes + 255) & ~(size_t)255; return p; };
  unsigned int*   xb    = (unsigned int*)  take((size_t)N * C1in * 2); // bf16 x, pair-packed
  unsigned int*   w1b   = (unsigned int*)  take((size_t)C1in * C1 * 2);
  unsigned int*   w2b   = (unsigned int*)  take((size_t)C1 * C2 * 2);
  float*          h1    = (float*)         take((size_t)N * C1 * 4);
  unsigned short* h1b   = (unsigned short*)take((size_t)N * C1 * 2);
  float*          agg1  = (float*)         take((size_t)N * C1 * 4);
  float*          as_   = (float*)         take((size_t)N * 4);
  float*          ad_   = (float*)         take((size_t)N * 4);
  float*          emax  = (float*)         take((size_t)N * 4);
  float*          denom = (float*)         take((size_t)N * 4);
  float*          ee    = (float*)         take((size_t)ET * 4);
  float*          h2    = (float*)xb;                               // alias
  (void)ws_size; (void)n_in; (void)in_sizes; (void)out_size;

  const int B = 256;
  const int aggBlocks = cdiv(ET, EDGES_PER_WAVE * 8);  // 8 waves per block

  // ---- layer 1 ----
  cvt_f32_bf16_kernel<<<cdiv(N * C1in, B), B, 0, stream>>>(x, (unsigned short*)xb, N * C1in);
  pack_w_kernel<<<cdiv(C1 * (C1in / 2), B), B, 0, stream>>>(W1, w1b, C1in, C1);
  pack_w_kernel<<<cdiv(C2 * (C1 / 2), B), B, 0, stream>>>(W2, w2b, C1, C2);

  dim3 g1(cdiv(N, 128), C1 / 64);
  gemm_bf16_wmma_kernel<<<g1, B, 0, stream>>>(xb, w1b, h1, N, C1in, C1);

  rowdot2_kernel<<<cdiv(N * 32, B), B, 0, stream>>>(h1, a1s, a1d, as_, ad_, N, C1);

  fill_f32_kernel<<<cdiv(N, B), B, 0, stream>>>(emax, -1e30f, N);
  fill_f32_kernel<<<cdiv(N, B), B, 0, stream>>>(denom, 0.f, N);
  bias_init_kernel<<<cdiv(N * C1, B), B, 0, stream>>>(agg1, b1, N * C1, C1);

  edge_max_kernel<<<cdiv(ET, B), B, 0, stream>>>(src, dst, as_, ad_, ee, emax, E, ET);
  edge_expsum_kernel<<<cdiv(ET, B), B, 0, stream>>>(src, dst, ee, emax, denom, E, ET);
  edge_agg_kernel<<<aggBlocks, B, 0, stream>>>(src, dst, ee, denom, h1, agg1, E, ET, C1);

  relu_bf16_kernel<<<cdiv(N * C1, B), B, 0, stream>>>(agg1, h1b, N * C1);

  // ---- layer 2 ----
  dim3 g2(cdiv(N, 128), C2 / 64);
  gemm_bf16_wmma_kernel<<<g2, B, 0, stream>>>((const unsigned int*)h1b, w2b, h2, N, C1, C2);

  rowdot2_kernel<<<cdiv(N * 32, B), B, 0, stream>>>(h2, a2s, a2d, as_, ad_, N, C2);

  fill_f32_kernel<<<cdiv(N, B), B, 0, stream>>>(emax, -1e30f, N);
  fill_f32_kernel<<<cdiv(N, B), B, 0, stream>>>(denom, 0.f, N);
  bias_init_kernel<<<cdiv(N * C2, B), B, 0, stream>>>(out, b2, N * C2, C2);

  edge_max_kernel<<<cdiv(ET, B), B, 0, stream>>>(src, dst, as_, ad_, ee, emax, E, ET);
  edge_expsum_kernel<<<cdiv(ET, B), B, 0, stream>>>(src, dst, ee, emax, denom, E, ET);
  edge_agg_kernel<<<aggBlocks, B, 0, stream>>>(src, dst, ee, denom, h2, out, E, ET, C2);
}
